// Model_46136538693975
// MI455X (gfx1250) — matrix operand order repaired and checked by reading, not compiled, not run on hardware
//
#include <hip/hip_runtime.h>

typedef unsigned short u16;
typedef unsigned long long u64;
typedef __attribute__((ext_vector_type(16))) __bf16 bf16x16;
typedef __attribute__((ext_vector_type(8)))  float   f32x8;

union FragU { uint4 q[2]; bf16x16 v; };

// ---- problem constants ----
#define NB   32          // batch
#define NS   1024        // seq len == codebook size
#define ND   512         // dim
#define NP   720         // pred len
#define NH   4           // heads
#define NC   1024        // codes
#define NBS  (NB*NS)     // 32768
#define NHD  (NH*ND)     // 2048
#define Y_ELEMS  (NB*NP*ND)     // 11796480
#define IDX_ELEMS (NBS*NH)      // 131072

// ---- workspace offsets (bytes, all 256-aligned) ----
#define OFF_X0B   ((size_t)0)                         // 32768*512 bf16
#define OFF_WINB  (OFF_X0B  + (size_t)NBS*ND*2)       // 2048*512 bf16
#define OFF_EMBB  (OFF_WINB + (size_t)NHD*ND*2)       // 4*1024*512 bf16
#define OFF_WOUTB (OFF_EMBB + (size_t)NH*NC*ND*2)     // 512*2048 bf16
#define OFF_WLINB (OFF_WOUTB+ (size_t)ND*NHD*2)       // 720*1024 bf16
#define OFF_E2    (OFF_WLINB+ (size_t)NP*NS*2)        // 4*1024 f32
#define OFF_NB    (OFF_E2   + (size_t)NH*NC*4)        // 4*1024*512 bf16 normalized
#define OFF_XI    (OFF_NB   + (size_t)NH*NC*ND*2)     // 32768*2048 bf16
#define OFF_QB    (OFF_XI   + (size_t)NBS*NHD*2)      // 32768*2048 bf16
#define OFF_QT    (OFF_QB   + (size_t)NBS*NHD*2)      // 32*512*1024 bf16 (quantized^T)
#define OFF_ARG   (OFF_QT   + (size_t)NB*ND*NS*2)     // 4*32768 u64 argmin keys
#define OFF_IND   (OFF_ARG  + (size_t)NH*NBS*8)       // 4*32768 i32
#define OFF_ACC   (OFF_IND  + (size_t)NH*NBS*4)       // [0]=commit, [1]=ortho

__device__ __forceinline__ u16 f2bf(float f) {
  unsigned u = __float_as_uint(f);
  unsigned r = u + 0x7FFFu + ((u >> 16) & 1u);
  return (u16)(r >> 16);
}
__device__ __forceinline__ float bf2f(u16 u) {
  return __uint_as_float(((unsigned)u) << 16);
}
__device__ __forceinline__ u64 pack_key(float s, unsigned c) {
  unsigned u = __float_as_uint(s);
  u = (u & 0x80000000u) ? ~u : (u | 0x80000000u);   // order-preserving map
  return ((u64)u << 32) | (u64)c;
}
__device__ __forceinline__ u64 shfl_min64(u64 k, int mask) {
  unsigned lo = (unsigned)k, hi = (unsigned)(k >> 32);
  unsigned olo = (unsigned)__shfl_xor((int)lo, mask, 32);
  unsigned ohi = (unsigned)__shfl_xor((int)hi, mask, 32);
  u64 o = ((u64)ohi << 32) | (u64)olo;
  return o < k ? o : k;
}

// async copy of 16B global -> LDS, tracked by ASYNCcnt
__device__ __forceinline__ void async_b128(const u16* g, u16* l) {
  unsigned lofs = (unsigned)(size_t)l;                 // low 32 bits = LDS offset
  unsigned long long ga = (unsigned long long)(size_t)g;
  asm volatile("global_load_async_to_lds_b128 %0, %1, off"
               :: "v"(lofs), "v"(ga) : "memory");
}
__device__ __forceinline__ void wait_async_le4() {
  asm volatile("s_wait_asynccnt 0x4" ::: "memory");
}
__device__ __forceinline__ void wait_async_0() {
  asm volatile("s_wait_asynccnt 0x0" ::: "memory");
}

// =====================================================================
// bf16 GEMM core: C[128x128] tile = A[M,K] * Bt[N,K]^T, fp32 accum.
// Block = 256 threads = 8 waves laid out 4 (M) x 2 (N); wave tile 32x64.
// v_wmma_f32_16x16x32_bf16 compute; double-buffered LDS fed by
// global_load_async_to_lds_b128 (ASYNCcnt), overlapping DMA with WMMA.
// OOB rows are clamped to the last valid row (outputs are store-guarded),
// keeping EXEC full as async/WMMA require.
// =====================================================================
__device__ __forceinline__ void gemm_core(
    const u16* __restrict__ A, int lda, int M, int m0,
    const u16* __restrict__ Bt, int ldb, int N, int n0,
    int K, f32x8 acc[2][4])
{
  __shared__ __align__(16) u16 sA[2][128 * 40];
  __shared__ __align__(16) u16 sB[2][128 * 40];

  const int tid  = threadIdx.x;
  const int lane = tid & 31;
  const int wave = tid >> 5;
  const int wm = wave & 3, wn = wave >> 2;
  const int hi = lane >> 4, l15 = lane & 15;

  // per-thread staging coordinates (2 segments each of A and B per tile)
  int rowi[2], segi[2], gri[2], gci[2];
  #pragma unroll
  for (int it = 0; it < 2; ++it) {
    int i = tid + it * 256;
    rowi[it] = i >> 2;
    segi[it] = i & 3;
    int gr = m0 + rowi[it]; if (gr >= M) gr = M - 1;   // clamp, store-guarded
    int gc = n0 + rowi[it]; if (gc >= N) gc = N - 1;
    gri[it] = gr; gci[it] = gc;
  }

  const f32x8 vz = {0.f,0.f,0.f,0.f,0.f,0.f,0.f,0.f};
  #pragma unroll
  for (int mi = 0; mi < 2; ++mi)
    #pragma unroll
    for (int ni = 0; ni < 4; ++ni) acc[mi][ni] = vz;

  const int nt = K >> 5;

  // prologue: issue tile 0 into buffer 0 (4 async b128 per wave)
  #pragma unroll
  for (int it = 0; it < 2; ++it) {
    async_b128(A + (size_t)gri[it] * lda + segi[it] * 8,
               &sA[0][rowi[it] * 40 + segi[it] * 8]);
    async_b128(Bt + (size_t)gci[it] * ldb + segi[it] * 8,
               &sB[0][rowi[it] * 40 + segi[it] * 8]);
  }

  for (int t = 0; t < nt; ++t) {
    const int cur = t & 1;
    if (t + 1 < nt) {
      const int kt1 = (t + 1) << 5;
      #pragma unroll
      for (int it = 0; it < 2; ++it) {
        async_b128(A + (size_t)gri[it] * lda + kt1 + segi[it] * 8,
                   &sA[cur ^ 1][rowi[it] * 40 + segi[it] * 8]);
        async_b128(Bt + (size_t)gci[it] * ldb + kt1 + segi[it] * 8,
                   &sB[cur ^ 1][rowi[it] * 40 + segi[it] * 8]);
      }
      wait_async_le4();   // in-order: current tile's 4 ops complete
    } else {
      wait_async_0();
    }
    __syncthreads();      // all waves' current tile resident

    FragU a[2];
    #pragma unroll
    for (int mi = 0; mi < 2; ++mi) {
      int r = wm * 32 + mi * 16 + l15;
      // A 16x32 bf16 layout: lanes 0-15 K{0..7,16..23}, lanes 16-31 K{8..15,24..31}
      a[mi].q[0] = *(const uint4*)(&sA[cur][r * 40 + hi * 8]);
      a[mi].q[1] = *(const uint4*)(&sA[cur][r * 40 + 16 + hi * 8]);
    }
    #pragma unroll
    for (int ni = 0; ni < 4; ++ni) {
      int c = wn * 64 + ni * 16 + l15;
      // B 32x16 bf16 layout: lanes 0-15 K{0..7,16..23}, lanes 16-31 K{8..15,24..31}
      FragU b;
      b.q[0] = *(const uint4*)(&sB[cur][c * 40 + hi * 8]);
      b.q[1] = *(const uint4*)(&sB[cur][c * 40 + 16 + hi * 8]);
      #pragma unroll
      for (int mi = 0; mi < 2; ++mi) {
        acc[mi][ni] = __builtin_amdgcn_wmma_f32_16x16x32_bf16(
            false, a[mi].v, false, b.v, (short)0, acc[mi][ni], false, false);
      }
    }
    __syncthreads();      // reads done before this buffer is overwritten (t+2)
  }
}

// ---- epilogue coordinate helpers ----
struct TC { int wm, wn, hi, l15; };
__device__ __forceinline__ TC tcoords() {
  int tid = threadIdx.x; int lane = tid & 31; int w = tid >> 5;
  TC t; t.wm = w & 3; t.wn = w >> 2; t.hi = lane >> 4; t.l15 = lane & 15;
  return t;
}

// =====================================================================
// Kernels
// =====================================================================
__global__ void k_init(u64* __restrict__ argbuf, float* __restrict__ accf) {
  int gid = blockIdx.x * 256 + threadIdx.x;
  if (gid < NH * NBS) argbuf[gid] = ~0ull;
  if (gid < 2) accf[gid] = 0.f;
}

// x0 = x - x[:, -1, :]  -> bf16
__global__ void k_x0(const float* __restrict__ x, u16* __restrict__ x0b) {
  int i = blockIdx.x * 256 + threadIdx.x;          // i < NBS*ND/4
  int e = i * 4;
  int row = e >> 9, d = e & 511;
  int b = row >> 10;
  const float4 xv = *(const float4*)(x + e);
  const float4 lv = *(const float4*)(x + (((size_t)((b << 10) | (NS - 1))) << 9) + d);
  unsigned p0 = (unsigned)f2bf(xv.x - lv.x) | ((unsigned)f2bf(xv.y - lv.y) << 16);
  unsigned p1 = (unsigned)f2bf(xv.z - lv.z) | ((unsigned)f2bf(xv.w - lv.w) << 16);
  *(uint2*)(x0b + e) = make_uint2(p0, p1);
}

__global__ void k_cvt(u16* __restrict__ dst, const float* __restrict__ src, int n4) {
  int i = blockIdx.x * 256 + threadIdx.x;
  if (i >= n4) return;
  int e = i * 4;
  const float4 v = *(const float4*)(src + e);
  unsigned p0 = (unsigned)f2bf(v.x) | ((unsigned)f2bf(v.y) << 16);
  unsigned p1 = (unsigned)f2bf(v.z) | ((unsigned)f2bf(v.w) << 16);
  *(uint2*)(dst + e) = make_uint2(p0, p1);
}

// per-code ||e||^2 and normalized bf16 codebook
__global__ void k_rownorm(const float* __restrict__ embed,
                          float* __restrict__ e2, u16* __restrict__ nb) {
  int row = blockIdx.x;                 // h*NC + c
  size_t base = (size_t)row * ND;
  float s = 0.f;
  for (int j = threadIdx.x; j < ND; j += 256) { float v = embed[base + j]; s += v * v; }
  __shared__ float red[256];
  red[threadIdx.x] = s; __syncthreads();
  for (int o = 128; o > 0; o >>= 1) { if (threadIdx.x < o) red[threadIdx.x] += red[threadIdx.x + o]; __syncthreads(); }
  __shared__ float tot;
  if (threadIdx.x == 0) { tot = red[0]; e2[row] = red[0]; }
  __syncthreads();
  float inv = rsqrtf(tot);
  for (int j = threadIdx.x; j < ND; j += 256)
    nb[base + j] = f2bf(embed[base + j] * inv);
}

// xi = x0 @ w_in^T + b_in     (A: x0b [NBS,ND], Bt: w_inb [NHD,ND])
__global__ void k_gemm_proj_in(const u16* __restrict__ A, const u16* __restrict__ Bt,
                               const float* __restrict__ b_in, u16* __restrict__ xi) {
  int m0 = blockIdx.y * 128, n0 = blockIdx.x * 128;
  f32x8 acc[2][4];
  gemm_core(A, ND, NBS, m0, Bt, ND, NHD, n0, ND, acc);
  TC t = tcoords();
  #pragma unroll
  for (int mi = 0; mi < 2; ++mi)
    #pragma unroll
    for (int ni = 0; ni < 4; ++ni) {
      int ng = n0 + t.wn * 64 + ni * 16 + t.l15;
      float bb = b_in[ng];
      #pragma unroll
      for (int r = 0; r < 8; ++r) {
        int mg = m0 + t.wm * 32 + mi * 16 + r + t.hi * 8;
        xi[(size_t)mg * NHD + ng] = f2bf(acc[mi][ni][r] + bb);
      }
    }
}

// distances: score = ||e||^2 - 2 f.e ; per-row argmin via u64 atomicMin
__global__ void k_gemm_dist(const u16* __restrict__ xi, const u16* __restrict__ embb,
                            const float* __restrict__ e2, u64* __restrict__ argbuf) {
  int h = blockIdx.z;
  int m0 = blockIdx.y * 128, n0 = blockIdx.x * 128;
  f32x8 acc[2][4];
  gemm_core(xi + h * ND, NHD, NBS, m0,
            embb + (size_t)h * NC * ND, ND, NC, n0, ND, acc);
  TC t = tcoords();
  const float* e2h = e2 + h * NC;
  u64 keys[16];
  #pragma unroll
  for (int i = 0; i < 16; ++i) keys[i] = ~0ull;
  #pragma unroll
  for (int mi = 0; mi < 2; ++mi)
    #pragma unroll
    for (int ni = 0; ni < 4; ++ni) {
      unsigned cg = (unsigned)(n0 + t.wn * 64 + ni * 16 + t.l15);
      float ev = e2h[cg];
      #pragma unroll
      for (int r = 0; r < 8; ++r) {
        u64 k = pack_key(ev - 2.f * acc[mi][ni][r], cg);
        int idx = mi * 8 + r;
        if (k < keys[idx]) keys[idx] = k;
      }
    }
  #pragma unroll
  for (int mask = 1; mask < 16; mask <<= 1)
    #pragma unroll
    for (int i = 0; i < 16; ++i) keys[i] = shfl_min64(keys[i], mask);
  if (t.l15 == 0) {
    u64* argh = argbuf + (size_t)h * NBS;
    #pragma unroll
    for (int mi = 0; mi < 2; ++mi)
      #pragma unroll
      for (int r = 0; r < 8; ++r) {
        int mg = m0 + t.wm * 32 + mi * 16 + r + t.hi * 8;
        atomicMin(&argh[mg], keys[mi * 8 + r]);
      }
  }
}

__global__ void k_indices(const u64* __restrict__ argbuf, int* __restrict__ indbuf,
                          int* __restrict__ idx_out) {
  int gid = blockIdx.x * 256 + threadIdx.x;
  if (gid >= NH * NBS) return;
  int c = (int)(unsigned)(argbuf[gid] & 0xffffffffull);
  indbuf[gid] = c;
  int h = gid >> 15, n = gid & (NBS - 1);
  idx_out[n * NH + h] = c;   // indices[b,s,h]
}

// gather q (bf16 into qb) + commitment loss accumulation
__global__ void k_commit(const float* __restrict__ embed, const u16* __restrict__ xi,
                         const int* __restrict__ indbuf, u16* __restrict__ qb,
                         float* __restrict__ accf) {
  int gid = blockIdx.x;            // h*NBS + n
  int h = gid >> 15, n = gid & (NBS - 1);
  int c = indbuf[gid];
  const float* ep = embed + ((size_t)(h * NC + c)) * ND;
  const u16*  fp = xi + (size_t)n * NHD + h * ND;
  u16*        qp = qb + (size_t)n * NHD + h * ND;
  float s = 0.f;
  for (int j = threadIdx.x; j < ND; j += 256) {
    float e = ep[j];
    float f = bf2f(fp[j]);
    float d = e - f;
    s += d * d;
    qp[j] = f2bf(e);
  }
  __shared__ float red[256];
  red[threadIdx.x] = s; __syncthreads();
  for (int o = 128; o > 0; o >>= 1) { if (threadIdx.x < o) red[threadIdx.x] += red[threadIdx.x + o]; __syncthreads(); }
  if (threadIdx.x == 0) atomicAdd(&accf[0], red[0]);
}

// ortho: cs = N N^T per head, accumulate sum(cs^2)
__global__ void k_gemm_cs(const u16* __restrict__ nb, float* __restrict__ accf) {
  int h = blockIdx.z;
  int m0 = blockIdx.y * 128, n0 = blockIdx.x * 128;
  const u16* Nh = nb + (size_t)h * NC * ND;
  f32x8 acc[2][4];
  gemm_core(Nh, ND, NC, m0, Nh, ND, NC, n0, ND, acc);
  float s = 0.f;
  #pragma unroll
  for (int mi = 0; mi < 2; ++mi)
    #pragma unroll
    for (int ni = 0; ni < 4; ++ni)
      #pragma unroll
      for (int r = 0; r < 8; ++r) { float v = acc[mi][ni][r]; s += v * v; }
  __shared__ float red[256];
  red[threadIdx.x] = s; __syncthreads();
  for (int o = 128; o > 0; o >>= 1) { if (threadIdx.x < o) red[threadIdx.x] += red[threadIdx.x + o]; __syncthreads(); }
  if (threadIdx.x == 0) atomicAdd(&accf[1], red[0]);
}

// quantized = qo @ w_out^T + b_out, stored transposed bf16: qT[b][d][s]
__global__ void k_gemm_proj_out(const u16* __restrict__ qb, const u16* __restrict__ Bt,
                                const float* __restrict__ b_out, u16* __restrict__ qT) {
  int m0 = blockIdx.y * 128, n0 = blockIdx.x * 128;
  f32x8 acc[2][4];
  gemm_core(qb, NHD, NBS, m0, Bt, NHD, ND, n0, NHD, acc);
  TC t = tcoords();
  #pragma unroll
  for (int mi = 0; mi < 2; ++mi)
    #pragma unroll
    for (int ni = 0; ni < 4; ++ni) {
      int ng = n0 + t.wn * 64 + ni * 16 + t.l15;      // d
      float bb = b_out[ng];
      #pragma unroll
      for (int r = 0; r < 8; ++r) {
        int mg = m0 + t.wm * 32 + mi * 16 + r + t.hi * 8;  // b*S+s
        int b = mg >> 10, sidx = mg & (NS - 1);
        qT[((size_t)(b * ND + ng)) * NS + sidx] = f2bf(acc[mi][ni][r] + bb);
      }
    }
}

// y[b] = w_lin @ quantized[b] + b_lin + seq_last[b]
__global__ void k_gemm_time(const u16* __restrict__ wlinb, const u16* __restrict__ qT,
                            const float* __restrict__ b_lin, const float* __restrict__ x,
                            float* __restrict__ y) {
  int b = blockIdx.z;
  int m0 = blockIdx.y * 128, n0 = blockIdx.x * 128;
  f32x8 acc[2][4];
  gemm_core(wlinb, NS, NP, m0, qT + (size_t)b * ND * NS, NS, ND, n0, NS, acc);
  TC t = tcoords();
  const float* last = x + (((size_t)b * NS + (NS - 1)) * ND);
  #pragma unroll
  for (int mi = 0; mi < 2; ++mi)
    #pragma unroll
    for (int ni = 0; ni < 4; ++ni) {
      int ng = n0 + t.wn * 64 + ni * 16 + t.l15;       // d
      float sl = last[ng];
      #pragma unroll
      for (int r = 0; r < 8; ++r) {
        int mg = m0 + t.wm * 32 + mi * 16 + r + t.hi * 8;   // p
        if (mg < NP)
          y[((size_t)(b * NP + mg)) * ND + ng] = acc[mi][ni][r] + b_lin[mg] + sl;
      }
    }
}

__global__ void k_final(const float* __restrict__ accf, float* __restrict__ out) {
  if (threadIdx.x == 0 && blockIdx.x == 0) {
    float commit = accf[0] / (float)((size_t)NH * NBS * ND);
    float ortho  = accf[1] / (float)(NH * NC * NC) - 1.0f / (float)NC;
    out[Y_ELEMS + IDX_ELEMS] = 1.0f * commit + 0.8f * ortho;
  }
}

// =====================================================================
extern "C" void kernel_launch(void* const* d_in, const int* in_sizes, int n_in,
                              void* d_out, int out_size, void* d_ws, size_t ws_size,
                              hipStream_t stream) {
  (void)in_sizes; (void)n_in; (void)out_size; (void)ws_size;
  const float* x     = (const float*)d_in[0];
  const float* w_in  = (const float*)d_in[1];
  const float* b_in  = (const float*)d_in[2];
  const float* embed = (const float*)d_in[3];
  const float* w_out = (const float*)d_in[4];
  const float* b_out = (const float*)d_in[5];
  const float* w_lin = (const float*)d_in[6];
  const float* b_lin = (const float*)d_in[7];
  float* out = (float*)d_out;

  char* ws = (char*)d_ws;
  u16*   x0b   = (u16*)(ws + OFF_X0B);
  u16*   winb  = (u16*)(ws + OFF_WINB);
  u16*   embb  = (u16*)(ws + OFF_EMBB);
  u16*   woutb = (u16*)(ws + OFF_WOUTB);
  u16*   wlinb = (u16*)(ws + OFF_WLINB);
  float* e2    = (float*)(ws + OFF_E2);
  u16*   nb    = (u16*)(ws + OFF_NB);
  u16*   xi    = (u16*)(ws + OFF_XI);
  u16*   qb    = (u16*)(ws + OFF_QB);
  u16*   qT    = (u16*)(ws + OFF_QT);
  u64*   argb  = (u64*)(ws + OFF_ARG);
  int*   indb  = (int*)(ws + OFF_IND);
  float* accf  = (float*)(ws + OFF_ACC);
  int*   idx_out = (int*)(out + Y_ELEMS);

  k_init<<<dim3(512), dim3(256), 0, stream>>>(argb, accf);
  k_x0<<<dim3(NBS * ND / 4 / 256), dim3(256), 0, stream>>>(x, x0b);
  k_cvt<<<dim3(NHD * ND / 4 / 256), dim3(256), 0, stream>>>(winb, w_in, NHD * ND / 4);
  k_cvt<<<dim3(NH * NC * ND / 4 / 256), dim3(256), 0, stream>>>(embb, embed, NH * NC * ND / 4);
  k_cvt<<<dim3(ND * NHD / 4 / 256), dim3(256), 0, stream>>>(woutb, w_out, ND * NHD / 4);
  k_cvt<<<dim3(NP * NS / 4 / 256), dim3(256), 0, stream>>>(wlinb, w_lin, NP * NS / 4);
  k_rownorm<<<dim3(NH * NC), dim3(256), 0, stream>>>(embed, e2, nb);

  k_gemm_proj_in<<<dim3(NHD / 128, NBS / 128), dim3(256), 0, stream>>>(x0b, winb, b_in, xi);
  k_gemm_dist<<<dim3(NC / 128, NBS / 128, NH), dim3(256), 0, stream>>>(xi, embb, e2, argb);
  k_indices<<<dim3(512), dim3(256), 0, stream>>>(argb, indb, idx_out);
  k_commit<<<dim3(NH * NBS), dim3(256), 0, stream>>>(embed, xi, indb, qb, accf);
  k_gemm_cs<<<dim3(NC / 128, NC / 128, NH), dim3(256), 0, stream>>>(nb, accf);
  k_gemm_proj_out<<<dim3(ND / 128, NBS / 128), dim3(256), 0, stream>>>(qb, woutb, b_out, qT);
  k_gemm_time<<<dim3(ND / 128, (NP + 127) / 128, NB), dim3(256), 0, stream>>>(wlinb, qT, b_lin, x, out);
  k_final<<<dim3(1), dim3(1), 0, stream>>>(accf, out);
}
